// LSTM_21002390078074
// MI455X (gfx1250) — matrix operand order repaired and checked by reading, not compiled, not run on hardware
//
#include <hip/hip_runtime.h>
#include <hip/hip_bf16.h>
#include <cstdint>
#include <cstddef>

// ---------------------------------------------------------------------------
// Types for CDNA5 WMMA intrinsics
// ---------------------------------------------------------------------------
typedef __bf16 v16bf __attribute__((ext_vector_type(16)));
typedef float  v8f   __attribute__((ext_vector_type(8)));
// Async-to-LDS builtin takes GCC-style int vectors (per hipcc diagnostic).
typedef int    v4i_vs __attribute__((__vector_size__(16)));

#define AS_GLOBAL __attribute__((address_space(1)))
#define AS_LDS    __attribute__((address_space(3)))

#if __has_builtin(__builtin_amdgcn_global_load_async_to_lds_b128)
#define HAS_ASYNC_LDS 1
#else
#define HAS_ASYNC_LDS 0
#endif

#define T_STEPS 128
#define BATCH   256
#define FDIM    1024

// Recurrence tiling: block = 128 threads = 4 waves; block tile 64(M) x 32(N).
// Wave positions: 4 m-tiles x 2 n-tiles; each wave owns an m-PAIR at one n,
// so every B fragment is reused by two WMMAs (1.5 ds_loads per WMMA).
#define BM   64
#define BN   32
#define KS   32     // K per WMMA (bf16)
#define KPAD 40     // padded LDS K stride (bf16 elems) to dodge bank conflicts

__device__ __forceinline__ __bf16 f2bf(float f) { return (__bf16)f; }

__device__ __forceinline__ float sigm(float x) { return 1.0f / (1.0f + __expf(-x)); }

// Branch-free tanh: avoids the OCML saveexec/cndmask path.
__device__ __forceinline__ float fast_tanh(float x) {
  float e = __expf(-2.0f * fabsf(x));
  float r = (1.0f - e) / (1.0f + e);
  return copysignf(r, x);
}

__device__ __forceinline__ void wait_async0() {
#if __has_builtin(__builtin_amdgcn_s_wait_asynccnt)
  __builtin_amdgcn_s_wait_asynccnt(0);
#elif defined(__AMDGCN__)
  asm volatile("s_wait_asynccnt 0x0" ::: "memory");
#endif
}

struct B32x2 { uint4 lo, hi; };

// Load a 16-element bf16 fragment as two ds_load_b128 (both 16B aligned).
__device__ __forceinline__ v16bf ld_frag(const __bf16* p, int off2_elems) {
  B32x2 t;
  t.lo = *(const uint4*)p;
  t.hi = *(const uint4*)(p + off2_elems);
  v16bf r;
  __builtin_memcpy(&r, &t, 32);
  return r;
}

// ---------------------------------------------------------------------------
// Weight packing fp32 -> bf16 (weights stay L2-resident afterwards)
// ---------------------------------------------------------------------------
__global__ void pack_bf16_kernel(const float* __restrict__ src,
                                 __bf16* __restrict__ dst, int n4) {
  int i = blockIdx.x * blockDim.x + threadIdx.x;
  if (i < n4) {
    float4 v = ((const float4*)src)[i];
    __bf16* d = dst + (size_t)i * 4;
    d[0] = f2bf(v.x); d[1] = f2bf(v.y); d[2] = f2bf(v.z); d[3] = f2bf(v.w);
  }
}

__global__ void zero_kernel(float* __restrict__ p, int n) {
  int i = blockIdx.x * blockDim.x + threadIdx.x;
  if (i < n) p[i] = 0.0f;
}

// ---------------------------------------------------------------------------
// One LSTM timestep: h_out = tanh(c) * sig(z Wo^T + bo),
//   c = sig(z Wf^T + bf) + sig(z Wi^T + bi) * tanh(z Wt^T + bt),
//   z = x_t + 2 h_in
// Four gate GEMMs fused; double-buffered LDS, one barrier per K-step;
// weights DMA'd straight to LDS via GLOBAL_LOAD_ASYNC_TO_LDS_B128.
// ---------------------------------------------------------------------------
__global__ __launch_bounds__(128) void lstm_step_kernel(
    const float* __restrict__ x_t, const float* __restrict__ h_in,
    float* __restrict__ h_out,
    const __bf16* __restrict__ Wf, const __bf16* __restrict__ Wi,
    const __bf16* __restrict__ Wt, const __bf16* __restrict__ Wo,
    const float* __restrict__ bf_, const float* __restrict__ bi_,
    const float* __restrict__ bt_, const float* __restrict__ bo_) {
  __shared__ alignas(16) __bf16 As[2][BM][KPAD];       // z tiles, bf16
  __shared__ alignas(16) __bf16 Bs[2][4][BN][KPAD];    // 4 gate weight tiles

  const int tid   = threadIdx.x;
  const int lane  = tid & 31;
  const int wave  = tid >> 5;           // 0..3
  const int wn    = wave & 1;           // n tile (0..1)
  const int wmB   = (wave >> 1) * 32;   // m pair base (0 or 32)
  const int brow0 = blockIdx.y * BM;
  const int ncol0 = blockIdx.x * BN;

  const v8f vzero = {0.f, 0.f, 0.f, 0.f, 0.f, 0.f, 0.f, 0.f};
  v8f acc0[4], acc1[4];
#pragma unroll
  for (int g = 0; g < 4; ++g) { acc0[g] = vzero; acc1[g] = vzero; }

  // B staging: one (gate, column) pair per thread; 32 bf16 = 64B per K-step.
  const int gB = tid >> 5;              // 0..3 (== wave)
  const int nB = tid & 31;              // 0..31
  const __bf16* wsel = (gB == 0) ? Wf : (gB == 1) ? Wi : (gB == 2) ? Wt : Wo;
  const __bf16* wptr = wsel + (size_t)(ncol0 + nB) * FDIM;

  // A staging: row m, 16 consecutive k per thread.
  const int mA  = tid >> 1;             // 0..63
  const int kkA = (tid & 1) << 4;       // 0 or 16
  const float* xptr = x_t + (size_t)(brow0 + mA) * FDIM + kkA;
  const float* hptr = h_in + (size_t)(brow0 + mA) * FDIM + kkA;

  // Fragment coordinates.
  const int am0 = wmB + (lane & 15);
  const int am1 = am0 + 16;
  const int akb = (lane >> 4) << 3;
  const int bn2 = wn * 16 + (lane & 15);
  const int bkb = (lane >> 4) << 3;

  // Pipeline registers (A only; B goes straight to LDS on the async path).
  float4 x0, x1, x2, x3, h0, h1, h2, h3;
#if !HAS_ASYNC_LDS
  uint4 w0, w1;
#endif

  auto stageB_issue = [&](int buf, int k0) {
#if HAS_ASYNC_LDS
    AS_GLOBAL v4i_vs* gp =
        (AS_GLOBAL v4i_vs*)(void*)const_cast<__bf16*>(wptr + k0);
    AS_LDS v4i_vs* lp = (AS_LDS v4i_vs*)(void*)&Bs[buf][gB][nB][0];
    __builtin_amdgcn_global_load_async_to_lds_b128(gp, lp, 0, 0);
    __builtin_amdgcn_global_load_async_to_lds_b128(gp, lp, 16, 0);
    __builtin_amdgcn_global_load_async_to_lds_b128(gp, lp, 32, 0);
    __builtin_amdgcn_global_load_async_to_lds_b128(gp, lp, 48, 0);
#else
    w0 = *(const uint4*)(wptr + k0);
    w1 = *(const uint4*)(wptr + k0 + 8);
    (void)buf;
#endif
  };

#if !HAS_ASYNC_LDS
  auto stageB_store = [&](int buf) {
    *(uint4*)&Bs[buf][gB][nB][0] = w0;
    *(uint4*)&Bs[buf][gB][nB][8] = w1;
  };
#endif

  auto gloadA = [&](int k0) {
    x0 = *(const float4*)(xptr + k0);
    x1 = *(const float4*)(xptr + k0 + 4);
    x2 = *(const float4*)(xptr + k0 + 8);
    x3 = *(const float4*)(xptr + k0 + 12);
    h0 = *(const float4*)(hptr + k0);
    h1 = *(const float4*)(hptr + k0 + 4);
    h2 = *(const float4*)(hptr + k0 + 8);
    h3 = *(const float4*)(hptr + k0 + 12);
  };

  auto sstoreA = [&](int buf) {
    __bf16* ar = &As[buf][mA][kkA];
    ar[0]  = f2bf(x0.x + 2.0f * h0.x);  ar[1]  = f2bf(x0.y + 2.0f * h0.y);
    ar[2]  = f2bf(x0.z + 2.0f * h0.z);  ar[3]  = f2bf(x0.w + 2.0f * h0.w);
    ar[4]  = f2bf(x1.x + 2.0f * h1.x);  ar[5]  = f2bf(x1.y + 2.0f * h1.y);
    ar[6]  = f2bf(x1.z + 2.0f * h1.z);  ar[7]  = f2bf(x1.w + 2.0f * h1.w);
    ar[8]  = f2bf(x2.x + 2.0f * h2.x);  ar[9]  = f2bf(x2.y + 2.0f * h2.y);
    ar[10] = f2bf(x2.z + 2.0f * h2.z);  ar[11] = f2bf(x2.w + 2.0f * h2.w);
    ar[12] = f2bf(x3.x + 2.0f * h3.x);  ar[13] = f2bf(x3.y + 2.0f * h3.y);
    ar[14] = f2bf(x3.z + 2.0f * h3.z);  ar[15] = f2bf(x3.w + 2.0f * h3.w);
  };

  auto compute = [&](int buf) {
    v16bf a0 = ld_frag(&As[buf][am0][akb], 16);
    v16bf a1 = ld_frag(&As[buf][am1][akb], 16);
#pragma unroll
    for (int g = 0; g < 4; ++g) {
      v16bf bfr = ld_frag(&Bs[buf][g][bn2][bkb], 16);
      acc0[g] = __builtin_amdgcn_wmma_f32_16x16x32_bf16(
          false, a0, false, bfr, (short)0, acc0[g], false, false);
      acc1[g] = __builtin_amdgcn_wmma_f32_16x16x32_bf16(
          false, a1, false, bfr, (short)0, acc1[g], false, false);
    }
  };

  // Software pipeline: one barrier per K-step; async B + prefetched A overlap WMMA.
  stageB_issue(0, 0);
  gloadA(0);
  for (int k0 = 0; k0 < FDIM; k0 += 2 * KS) {
    sstoreA(0);
#if !HAS_ASYNC_LDS
    stageB_store(0);
#endif
    wait_async0();
    __syncthreads();
    stageB_issue(1, k0 + KS);            // FDIM/KS is even: always in-bounds
    gloadA(k0 + KS);
    compute(0);
    sstoreA(1);
#if !HAS_ASYNC_LDS
    stageB_store(1);
#endif
    wait_async0();
    __syncthreads();
    if (k0 + 2 * KS < FDIM) {
      stageB_issue(0, k0 + 2 * KS);
      gloadA(k0 + 2 * KS);
    }
    compute(1);
  }

  // Epilogue: bias + gate nonlinearities + h update (two m tiles).
  const int ncol  = ncol0 + wn * 16 + (lane & 15);
  const float bfv = bf_[ncol], biv = bi_[ncol], btv = bt_[ncol], bov = bo_[ncol];
  const int rb0 = brow0 + wmB + ((lane >> 4) << 3);
#pragma unroll
  for (int r = 0; r < 8; ++r) {
    float fg = sigm(acc0[0][r] + bfv);
    float ig = sigm(acc0[1][r] + biv);
    float tg = fast_tanh(acc0[2][r] + btv);
    float og = sigm(acc0[3][r] + bov);
    float c  = fg + ig * tg;
    h_out[(size_t)(rb0 + r) * FDIM + ncol] = fast_tanh(c) * og;
  }
  const int rb1 = rb0 + 16;
#pragma unroll
  for (int r = 0; r < 8; ++r) {
    float fg = sigm(acc1[0][r] + bfv);
    float ig = sigm(acc1[1][r] + biv);
    float tg = fast_tanh(acc1[2][r] + btv);
    float og = sigm(acc1[3][r] + bov);
    float c  = fg + ig * tg;
    h_out[(size_t)(rb1 + r) * FDIM + ncol] = fast_tanh(c) * og;
  }
}

// ---------------------------------------------------------------------------
// Head GEMM: C[M=256,N] = A[256,K] @ W^T + bias  (A fp32 -> bf16 in LDS)
// Block = 256 threads = 8 waves; tile 64x32; one 16x16 position per wave.
// ---------------------------------------------------------------------------
__global__ __launch_bounds__(256) void gemm_bias_kernel(
    const float* __restrict__ A, const __bf16* __restrict__ W,
    const float* __restrict__ bias, float* __restrict__ C, int N, int K) {
  __shared__ alignas(16) __bf16 As[BM][KPAD];
  __shared__ alignas(16) __bf16 Bs[BN][KPAD];

  const int tid   = threadIdx.x;
  const int lane  = tid & 31;
  const int wave  = tid >> 5;
  const int wm    = wave >> 1;
  const int wn    = wave & 1;
  const int brow0 = blockIdx.y * BM;
  const int ncol0 = blockIdx.x * BN;

  const v8f vzero = {0.f, 0.f, 0.f, 0.f, 0.f, 0.f, 0.f, 0.f};
  v8f acc = vzero;

  const int mA  = tid >> 2;
  const int kkA = (tid & 3) << 3;
  const float* aptr = A + (size_t)(brow0 + mA) * K + kkA;

  const int nBcol = tid >> 3;           // 0..31
  const int kkB   = (tid & 7) << 2;     // 0,4,...,28
  const __bf16* wptr = W + (size_t)(ncol0 + nBcol) * K + kkB;

  for (int k0 = 0; k0 < K; k0 += KS) {
    __syncthreads();
    {
      float4 a0 = *(const float4*)(aptr + k0);
      float4 a1 = *(const float4*)(aptr + k0 + 4);
      __bf16* ar = &As[mA][kkA];
      ar[0] = f2bf(a0.x); ar[1] = f2bf(a0.y);
      ar[2] = f2bf(a0.z); ar[3] = f2bf(a0.w);
      ar[4] = f2bf(a1.x); ar[5] = f2bf(a1.y);
      ar[6] = f2bf(a1.z); ar[7] = f2bf(a1.w);
    }
    {
      uint2 v = *(const uint2*)(wptr + k0);
      *(uint2*)&Bs[nBcol][kkB] = v;
    }
    __syncthreads();

    const int am  = wm * 16 + (lane & 15);
    const int akb = (lane >> 4) << 3;
    v16bf afrag = ld_frag(&As[am][akb], 16);

    const int bn2 = wn * 16 + (lane & 15);
    const int bkb = (lane >> 4) << 3;
    v16bf bfrag = ld_frag(&Bs[bn2][bkb], 16);

    acc = __builtin_amdgcn_wmma_f32_16x16x32_bf16(
        false, afrag, false, bfrag, (short)0, acc, false, false);
  }

  const int ncol  = ncol0 + wn * 16 + (lane & 15);
  const float bv  = bias[ncol];
  const int rbase = brow0 + wm * 16 + ((lane >> 4) << 3);
#pragma unroll
  for (int r = 0; r < 8; ++r)
    C[(size_t)(rbase + r) * N + ncol] = acc[r] + bv;
}

// ---------------------------------------------------------------------------
// Training-mode BatchNorm (batch stats over M=256 rows) + ReLU, in place.
// ---------------------------------------------------------------------------
__global__ __launch_bounds__(256) void bn_relu_kernel(
    float* __restrict__ X, const float* __restrict__ g,
    const float* __restrict__ b, int N) {
  const int col = blockIdx.x;
  const int t   = threadIdx.x;
  float x = X[(size_t)t * N + col];
  __shared__ float s1[256];
  __shared__ float s2[256];
  s1[t] = x;
  s2[t] = x * x;
  __syncthreads();
  for (int s = 128; s > 0; s >>= 1) {
    if (t < s) { s1[t] += s1[t + s]; s2[t] += s2[t + s]; }
    __syncthreads();
  }
  float mu  = s1[0] * (1.0f / 256.0f);
  float var = s2[0] * (1.0f / 256.0f) - mu * mu;
  float y   = g[col] * (x - mu) * rsqrtf(var + 1e-5f) + b[col];
  X[(size_t)t * N + col] = fmaxf(y, 0.0f);
}

// ---------------------------------------------------------------------------
// Final projection: out[256,2] = concat(X3[256,256], sent[256,1]) @ Wout^T + bout
// ---------------------------------------------------------------------------
__global__ __launch_bounds__(256) void final_kernel(
    const float* __restrict__ X, const float* __restrict__ sent,
    const float* __restrict__ Wout, const float* __restrict__ bout,
    float* __restrict__ out) {
  int i = blockIdx.x * blockDim.x + threadIdx.x;
  if (i >= BATCH * 2) return;
  int row = i >> 1, o = i & 1;
  const float* wr = Wout + o * 257;
  const float* xr = X + (size_t)row * 256;
  float s = bout[o] + sent[row] * wr[256];
#pragma unroll 8
  for (int k = 0; k < 256; ++k) s += xr[k] * wr[k];
  out[i] = s;
}

// ---------------------------------------------------------------------------
// Launcher
// ---------------------------------------------------------------------------
extern "C" void kernel_launch(void* const* d_in, const int* in_sizes, int n_in,
                              void* d_out, int out_size, void* d_ws, size_t ws_size,
                              hipStream_t stream) {
  const float* inputs = (const float*)d_in[0];
  const float* sent   = (const float*)d_in[1];
  const float* Wf  = (const float*)d_in[2];  const float* bfb = (const float*)d_in[3];
  const float* Wis = (const float*)d_in[4];  const float* bis = (const float*)d_in[5];
  const float* Wit = (const float*)d_in[6];  const float* bit_ = (const float*)d_in[7];
  const float* Wos = (const float*)d_in[8];  const float* bos = (const float*)d_in[9];
  const float* W1  = (const float*)d_in[10]; const float* b1  = (const float*)d_in[11];
  const float* g1  = (const float*)d_in[12]; const float* be1 = (const float*)d_in[13];
  const float* W2  = (const float*)d_in[14]; const float* b2  = (const float*)d_in[15];
  const float* g2  = (const float*)d_in[16]; const float* be2 = (const float*)d_in[17];
  const float* W3  = (const float*)d_in[18]; const float* b3  = (const float*)d_in[19];
  const float* g3  = (const float*)d_in[20]; const float* be3 = (const float*)d_in[21];
  const float* Wout = (const float*)d_in[22]; const float* bout = (const float*)d_in[23];

  char* ws = (char*)d_ws;
  size_t off = 0;
  auto take = [&](size_t bytes) -> void* {
    void* p = ws + off;
    off = (off + bytes + 255) & ~(size_t)255;
    return p;
  };

  __bf16* wf16  = (__bf16*)take((size_t)FDIM * FDIM * 2);
  __bf16* wis16 = (__bf16*)take((size_t)FDIM * FDIM * 2);
  __bf16* wit16 = (__bf16*)take((size_t)FDIM * FDIM * 2);
  __bf16* wos16 = (__bf16*)take((size_t)FDIM * FDIM * 2);
  __bf16* w1b   = (__bf16*)take((size_t)FDIM * FDIM * 2);
  __bf16* w2b   = (__bf16*)take((size_t)(FDIM / 2) * FDIM * 2);
  __bf16* w3b   = (__bf16*)take((size_t)(FDIM / 4) * (FDIM / 2) * 2);
  float*  hA    = (float*)take((size_t)BATCH * FDIM * 4);
  float*  hB    = (float*)take((size_t)BATCH * FDIM * 4);
  float*  t1    = (float*)take((size_t)BATCH * FDIM * 4);
  float*  t2    = (float*)take((size_t)BATCH * (FDIM / 2) * 4);
  float*  t3    = (float*)take((size_t)BATCH * (FDIM / 4) * 4);

  auto pack = [&](const float* s, __bf16* d, int n) {
    int n4 = n / 4;
    pack_bf16_kernel<<<(n4 + 255) / 256, 256, 0, stream>>>(s, d, n4);
  };
  pack(Wf,  wf16,  FDIM * FDIM);
  pack(Wis, wis16, FDIM * FDIM);
  pack(Wit, wit16, FDIM * FDIM);
  pack(Wos, wos16, FDIM * FDIM);
  pack(W1,  w1b,   FDIM * FDIM);
  pack(W2,  w2b,   (FDIM / 2) * FDIM);
  pack(W3,  w3b,   (FDIM / 4) * (FDIM / 2));

  zero_kernel<<<(BATCH * FDIM + 255) / 256, 256, 0, stream>>>(hA, BATCH * FDIM);

  // Sequential recurrence: one fused 4-gate GEMM kernel per timestep.
  dim3 sgrid(FDIM / BN, BATCH / BM);
  float* hc = hA;
  float* hn = hB;
  for (int t = 0; t < T_STEPS; ++t) {
    lstm_step_kernel<<<sgrid, 128, 0, stream>>>(
        inputs + (size_t)t * BATCH * FDIM, hc, hn,
        wf16, wis16, wit16, wos16, bfb, bis, bit_, bos);
    float* tmp = hc; hc = hn; hn = tmp;
  }

  // Head: Linear -> BN -> ReLU (x3), concat sentiment, final 257->2.
  gemm_bias_kernel<<<dim3(FDIM / BN, BATCH / BM), 256, 0, stream>>>(
      hc, w1b, b1, t1, FDIM, FDIM);
  bn_relu_kernel<<<FDIM, 256, 0, stream>>>(t1, g1, be1, FDIM);

  gemm_bias_kernel<<<dim3((FDIM / 2) / BN, BATCH / BM), 256, 0, stream>>>(
      t1, w2b, b2, t2, FDIM / 2, FDIM);
  bn_relu_kernel<<<FDIM / 2, 256, 0, stream>>>(t2, g2, be2, FDIM / 2);

  gemm_bias_kernel<<<dim3((FDIM / 4) / BN, BATCH / BM), 256, 0, stream>>>(
      t2, w3b, b3, t3, FDIM / 4, FDIM / 2);
  bn_relu_kernel<<<FDIM / 4, 256, 0, stream>>>(t3, g3, be3, FDIM / 4);

  final_kernel<<<2, 256, 0, stream>>>(t3, sent, Wout, bout, (float*)d_out);
}
